// CRF_3633542332437
// MI455X (gfx1250) — matrix operand order repaired and checked by reading, not compiled, not run on hardware
//
#include <hip/hip_runtime.h>

// CRF mean-field: x <- x0 + w * SeparableGaussBlur(softmax_C(x)), 5 iterations.
// Convs run as banded matmuls on v_wmma_f32_16x16x32_f16. Band matrices are
// materialized once per workgroup in LDS so all WMMA operands load as ds_load_b128.

typedef __attribute__((ext_vector_type(16))) _Float16 v16h;
typedef __attribute__((ext_vector_type(8)))  _Float16 h8;
typedef __attribute__((ext_vector_type(8)))  float    v8f;

#define NCLASS 19
#define HWDIM  512
#define PLANE  (512*512)
#define QS     40          // LDS row stride in halfs (80B): conflict-free 16B accesses

// ---------------- Kernel 1: channel softmax, f32 -> f16 q ----------------
__global__ void __launch_bounds__(256)
crf_softmax_kernel(const float* __restrict__ x, _Float16* __restrict__ q) {
    int p  = blockIdx.x * blockDim.x + threadIdx.x;   // 0 .. 8*PLANE-1
    int b  = p >> 18;                                 // / PLANE
    int hw = p & (PLANE - 1);
    const float* xp = x + (size_t)b * NCLASS * PLANE + hw;
    float v[NCLASS];
    float m = -1e30f;
#pragma unroll
    for (int c = 0; c < NCLASS; ++c) { v[c] = xp[(size_t)c * PLANE]; m = fmaxf(m, v[c]); }
    float s = 0.0f;
#pragma unroll
    for (int c = 0; c < NCLASS; ++c) { v[c] = __expf(v[c] - m); s += v[c]; }
    float inv = 1.0f / s;
    _Float16* qp = q + (size_t)b * NCLASS * PLANE + hw;
#pragma unroll
    for (int c = 0; c < NCLASS; ++c) qp[(size_t)c * PLANE] = (_Float16)(v[c] * inv);
}

// ---------------- Kernel 2: separable blur via WMMA + update ----------------
// One wave owns one 16x16 output tile of one (b,c) plane.
// Stage 1 (W conv): Out_h(32x16) = Q(32x32) x Kw(32x16)    -> 2 WMMAs, Q in A slot
// Stage 2 (H conv): Out(16x16)   = Kv(16x32) x Out_h(32x16) -> 1 WMMA, Kv in A slot
__global__ void __launch_bounds__(128)
crf_blur_update_kernel(const _Float16* __restrict__ q, const float* __restrict__ x0,
                       const float* __restrict__ spacings, const float* __restrict__ wgt,
                       const float* __restrict__ itheta, float* __restrict__ out) {
    __shared__ __align__(16) _Float16 qt   [4][32 * QS]; // haloed q tile (zero padded), wave-private
    __shared__ __align__(16) _Float16 mid  [4][16 * QS]; // Out_h TRANSPOSED [col][row], wave-private
    __shared__ __align__(16) _Float16 bandW[16 * QS];    // bandW[w][s] = kw[s-w]  (B-layout reads)
    __shared__ __align__(16) _Float16 bandV[16 * QS];    // bandV[h][r] = kh[r-h]  (A-layout reads)

    int tid  = threadIdx.x;
    int lane = tid & 31;
    int wv   = tid >> 5;
    int tile = blockIdx.x * 4 + wv;        // 0..1023 (32x32 tiles of 16x16)
    int tr   = tile >> 5;
    int tc   = tile & 31;
    int ch   = blockIdx.y;
    int b    = blockIdx.z;

    // ---- Build band matrices once per workgroup (branch-free tap generation) ----
    if (tid < 32) {
        int  rsel = tid & 15;
        bool isW  = tid < 16;
        float sp  = isW ? spacings[b * 2 + 1] : spacings[b * 2 + 0];
        float it  = isW ? itheta[1]           : itheta[0];
        _Float16* row = (isW ? bandW : bandV) + rsel * QS;
        h8 zz = {};
        h8* rv = (h8*)row;
#pragma unroll
        for (int j = 0; j < 5; ++j) rv[j] = zz;           // zero the 40-half row
#pragma unroll
        for (int d = 0; d <= 10; ++d) {                   // d is unroll-constant: no branches
            if (d != 5) {                                 // center tap excluded (zeroed)
                float t = (float)(d - 5) * sp * it;
                row[rsel + d] = (_Float16)__expf(-0.5f * t * t);
            }
        }
    }

    // ---- Load 26x26 halo region into zeroed 32x32 region (coalesced rows) ----
    const _Float16* qp = q + (size_t)(b * NCLASS + ch) * PLANE;
#pragma unroll
    for (int r = 0; r < 32; ++r) {                        // r is unroll-constant
        int gr = tr * 16 + r - 5;
        int gc = tc * 16 + lane - 5;
        _Float16 val = (_Float16)0.0f;
        if (r < 26 && lane < 26 && gr >= 0 && gr < HWDIM && gc >= 0 && gc < HWDIM)
            val = qp[gr * HWDIM + gc];
        qt[wv][r * QS + lane] = val;
    }
    __syncthreads();                                      // bands are block-shared

    int M     = lane & 15;              // row (A) / column (B,C,D) owned by this lane
    int hi16  = (lane >= 16);
    int abase = hi16 ? 8 : 0;           // table-order base, unused (A is read via hb)
    int kb    = hi16 ? 16 : 0;          // A and B k order used here: K 0-15 vs 16-31
    int rbase = hi16 ? 8 : 0;           // C/D layout: rows v / v+8
    int ha    = abase >> 3;             // h8 index within a row: 0 or 1
    int hb    = kb >> 3;                // h8 index within a row: 0 or 2

    // ---- Stage 1: A = q-tile rows, B = Kw band rows (all contiguous b128 reads) ----
    const h8* qrowT = (const h8*)&qt[wv][M * QS];
    const h8* qrowB = (const h8*)&qt[wv][(16 + M) * QS];
    h8 t0 = qrowT[hb], t1 = qrowT[hb + 1];
    h8 u0 = qrowB[hb], u1 = qrowB[hb + 1];
    const h8* bwr = (const h8*)&bandW[M * QS];
    h8 w0 = bwr[hb], w1 = bwr[hb + 1];

    v16h aT, aB, bk;
#pragma unroll
    for (int i = 0; i < 8; ++i) { aT[i] = t0[i]; aT[8 + i] = t1[i]; }
#pragma unroll
    for (int i = 0; i < 8; ++i) { aB[i] = u0[i]; aB[8 + i] = u1[i]; }
#pragma unroll
    for (int i = 0; i < 8; ++i) { bk[i] = w0[i]; bk[8 + i] = w1[i]; }

    v8f z = {};
    v8f accT = __builtin_amdgcn_wmma_f32_16x16x32_f16(false, aT, false, bk, (short)0, z, false, false);
    v8f accB = __builtin_amdgcn_wmma_f32_16x16x32_f16(false, aB, false, bk, (short)0, z, false, false);

    // Restage Out_h transposed: mid[col][row]; 16B packed stores (same-wave DS ordering).
    h8 p0, p1;
#pragma unroll
    for (int i = 0; i < 8; ++i) { p0[i] = (_Float16)accT[i]; p1[i] = (_Float16)accB[i]; }
    *(h8*)&mid[wv][M * QS + rbase]      = p0;   // rows rbase..rbase+7
    *(h8*)&mid[wv][M * QS + 16 + rbase] = p1;   // rows 16+rbase..16+rbase+7

    // ---- Stage 2: A = Kv band rows, B = Out_h columns (contiguous b128 reads) ----
    const h8* bvr = (const h8*)&bandV[M * QS];
    h8 a0 = bvr[hb], a1 = bvr[hb + 1];
    const h8* mrow = (const h8*)&mid[wv][M * QS];
    h8 m0 = mrow[hb], m1 = mrow[hb + 1];

    v16h a2, b2;
#pragma unroll
    for (int i = 0; i < 8; ++i) { a2[i] = a0[i]; a2[8 + i] = a1[i]; }
#pragma unroll
    for (int i = 0; i < 8; ++i) { b2[i] = m0[i]; b2[8 + i] = m1[i]; }

    v8f accF = __builtin_amdgcn_wmma_f32_16x16x32_f16(false, a2, false, b2, (short)0, z, false, false);

    // x_new = x0 + w * blur  (comp = -I makes the message update additive)
    float w = wgt[0];
    const float* xp = x0  + (size_t)(b * NCLASS + ch) * PLANE;
    float*       op = out + (size_t)(b * NCLASS + ch) * PLANE;
#pragma unroll
    for (int v = 0; v < 8; ++v) {
        int gh  = tr * 16 + rbase + v;
        int gw  = tc * 16 + M;
        int idx = gh * HWDIM + gw;
        op[idx] = xp[idx] + w * accF[v];
    }
}

extern "C" void kernel_launch(void* const* d_in, const int* in_sizes, int n_in,
                              void* d_out, int out_size, void* d_ws, size_t ws_size,
                              hipStream_t stream) {
    (void)in_sizes; (void)n_in; (void)out_size; (void)ws_size;
    const float* x_in     = (const float*)d_in[0];   // (8,19,512,512) f32
    const float* spacings = (const float*)d_in[1];   // (8,2) f32
    const float* wgt      = (const float*)d_in[2];   // scalar f32
    const float* itheta   = (const float*)d_in[3];   // (2,) f32
    float*       out      = (float*)d_out;           // (8,19,512,512) f32
    _Float16*    qbuf     = (_Float16*)d_ws;         // ~80 MB f16 scratch

    dim3 sgrid((8 * PLANE) / 256);
    dim3 bgrid(256, NCLASS, 8);                      // 1024 tiles / 4 waves, 19 ch, 8 batch

    for (int it = 0; it < 5; ++it) {
        const float* xcur = (it == 0) ? x_in : out;  // d_out doubles as x ping buffer
        crf_softmax_kernel<<<sgrid, 256, 0, stream>>>(xcur, qbuf);
        crf_blur_update_kernel<<<bgrid, 128, 0, stream>>>(qbuf, x_in, spacings, wgt, itheta, out);
    }
}
